// SwinTransformer_60155311948573
// MI455X (gfx1250) — matrix operand order repaired and checked by reading, not compiled, not run on hardware
//
#include <hip/hip_runtime.h>
#include <hip/hip_bf16.h>
#include <math.h>

typedef __attribute__((ext_vector_type(16))) _Float16 v16h;
typedef __attribute__((ext_vector_type(8)))  _Float16 h8;
typedef __attribute__((ext_vector_type(8)))  float    v8f;
typedef __attribute__((ext_vector_type(4)))  int      v4i;

#define LN_EPS 1e-5f
#define ANMAX 352   // 343 padded to multiple of 32

// CDNA5 async global->LDS path (builtin confirmed present on this toolchain;
// signature: (v4i addrspace(1)*, v4i addrspace(3)*, imm offset, imm cpol))
#if defined(__has_builtin)
#if __has_builtin(__builtin_amdgcn_global_load_async_to_lds_b128) && \
    __has_builtin(__builtin_amdgcn_s_wait_asynccnt)
#define USE_ASYNC_LDS 1
#endif
#endif

// ---- WMMA fragment index helpers (CDNA5 16x16x32 f16 layouts, CDNA5 ISA) ----
// A (16x32, MxK): lane<16: M=lane, K={0..7,16..23}; lane>=16: M=lane-16, K={8..15,24..31}
__device__ __forceinline__ int a_kidx(int lane, int j) {
  return (lane < 16) ? ((j < 8) ? j : j + 8) : ((j < 8) ? j + 8 : j + 16);
}
// B (32x16, KxN), this helper's contiguous k, not the ISA order: lane<16: N=lane, K=j; lane>=16: N=lane-16, K=j+16
__device__ __forceinline__ int b_kidx(int lane, int j) {
  return (lane < 16) ? j : (j + 16);
}
// C/D (16x16 f32): VGPR r: lanes0-15 -> M=r, lanes16-31 -> M=r+8; N = lane%16

// =============================== elementwise kernels ===============================

__global__ void k_patch_embed(const float* __restrict__ in, const float* __restrict__ w,
                              const float* __restrict__ bias, float* __restrict__ out,
                              int B, int D, int H, int W, int C) {
  long idx = (long)blockIdx.x * blockDim.x + threadIdx.x;
  long total = (long)B * D * H * W * C;
  if (idx >= total) return;
  int c = (int)(idx % C);
  long v = idx / C;
  int x = (int)(v % W); v /= W;
  int y = (int)(v % H); v /= H;
  int z = (int)(v % D); v /= D;
  int b = (int)v;
  int D2 = 2 * D, H2 = 2 * H, W2 = 2 * W;
  float acc = bias[c];
#pragma unroll
  for (int dz = 0; dz < 2; ++dz)
#pragma unroll
    for (int dy = 0; dy < 2; ++dy)
#pragma unroll
      for (int dx = 0; dx < 2; ++dx) {
        float iv = in[(((long)b * D2 + (2 * z + dz)) * H2 + (2 * y + dy)) * W2 + (2 * x + dx)];
        acc += iv * w[((c * 2 + dz) * 2 + dy) * 2 + dx];
      }
  out[idx] = acc;
}

// src channels-last (B,D,H,W,C) -> dst NCDHW normalized over C
__global__ void k_proj_norm_out(const float* __restrict__ src, float* __restrict__ dst,
                                int B, int C, int D, int H, int W) {
  long v = (long)blockIdx.x * blockDim.x + threadIdx.x;
  long nvox = (long)B * D * H * W;
  if (v >= nvox) return;
  const float* r = src + v * C;
  float mu = 0.f;
  for (int c = 0; c < C; ++c) mu += r[c];
  mu /= (float)C;
  float var = 0.f;
  for (int c = 0; c < C; ++c) { float d = r[c] - mu; var += d * d; }
  var /= (float)C;
  float inv = rsqrtf(var + LN_EPS);
  long spat = (long)D * H * W;
  long sp = v % spat;
  int b = (int)(v / spat);
  for (int c = 0; c < C; ++c)
    dst[((long)b * C + c) * spat + sp] = (r[c] - mu) * inv;
}

// layernorm, f32 in -> f16 out (all LN consumers are WMMA GEMM A-operands)
__global__ void k_layernorm(const float* __restrict__ src, const float* __restrict__ g,
                            const float* __restrict__ bb, _Float16* __restrict__ dst,
                            long rows, int C) {
  long v = (long)blockIdx.x * blockDim.x + threadIdx.x;
  if (v >= rows) return;
  const float* r = src + v * C;
  float mu = 0.f;
  for (int c = 0; c < C; ++c) mu += r[c];
  mu /= (float)C;
  float var = 0.f;
  for (int c = 0; c < C; ++c) { float d = r[c] - mu; var += d * d; }
  var /= (float)C;
  float inv = rsqrtf(var + LN_EPS);
  _Float16* o = dst + v * C;
  for (int c = 0; c < C; ++c) o[c] = (_Float16)((r[c] - mu) * inv * g[c] + bb[c]);
}

__global__ void k_cvt_f16(const float* __restrict__ src, _Float16* __restrict__ dst, long n) {
  long i = (long)blockIdx.x * blockDim.x + threadIdx.x;
  if (i < n) dst[i] = (_Float16)src[i];
}

// pad + roll(-shift) + window_partition; cubic dims; f16 passthrough
__global__ void k_win_gather(const _Float16* __restrict__ xn, _Float16* __restrict__ win,
                             int B, int D, int C, int wsz, int nwd, int Sp, int shift) {
  int nseq = wsz * wsz * wsz;
  int nw = nwd * nwd * nwd;
  long idx = (long)blockIdx.x * blockDim.x + threadIdx.x;
  long total = (long)B * nw * nseq * C;
  if (idx >= total) return;
  int c = (int)(idx % C);
  long t = idx / C;
  int e = (int)(t % nseq);
  long bw = t / nseq;
  int wlin = (int)(bw % nw);
  int b = (int)(bw / nw);
  int wz = wlin / (nwd * nwd), wy = (wlin / nwd) % nwd, wx = wlin % nwd;
  int ez = e / (wsz * wsz), ey = (e / wsz) % wsz, ex = e % wsz;
  int pz = wz * wsz + ez, py = wy * wsz + ey, px = wx * wsz + ex;
  int qz = (pz + shift) % Sp, qy = (py + shift) % Sp, qx = (px + shift) % Sp;
  _Float16 v = (_Float16)0.f;
  if (qz < D && qy < D && qx < D)
    v = xn[((((long)b * D + qz) * D + qy) * D + qx) * C + c];
  win[idx] = v;
}

// window_reverse + roll(+shift) + crop + residual add (in place on x), f32
__global__ void k_win_scatter_add(const float* __restrict__ proj, float* __restrict__ x,
                                  int B, int D, int C, int wsz, int nwd, int Sp, int shift) {
  int nseq = wsz * wsz * wsz;
  int nw = nwd * nwd * nwd;
  long idx = (long)blockIdx.x * blockDim.x + threadIdx.x;
  long total = (long)B * D * D * D * C;
  if (idx >= total) return;
  int c = (int)(idx % C);
  long v = idx / C;
  int px = (int)(v % D); v /= D;
  int py = (int)(v % D); v /= D;
  int pz = (int)(v % D); v /= D;
  int b = (int)v;
  int qz = (pz - shift + Sp) % Sp, qy = (py - shift + Sp) % Sp, qx = (px - shift + Sp) % Sp;
  int wz = qz / wsz, ez = qz % wsz;
  int wy = qy / wsz, ey = qy % wsz;
  int wx = qx / wsz, ex = qx % wsz;
  int wlin = (wz * nwd + wy) * nwd + wx;
  long tok = ((long)b * nw + wlin) * nseq + ((ez * wsz + ey) * wsz + ex);
  x[idx] += proj[tok * C + c];
}

// patch merging gather (replicates reference's x5/x6 duplication), f32
__global__ void k_merge_gather(const float* __restrict__ x, float* __restrict__ tmp,
                               int B, int D2, int C) {
  const int od[8] = {0, 1, 0, 0, 1, 0, 0, 1};
  const int oh[8] = {0, 0, 1, 0, 0, 1, 0, 1};
  const int ow[8] = {0, 0, 0, 1, 1, 0, 1, 1};
  long idx = (long)blockIdx.x * blockDim.x + threadIdx.x;
  int C8 = 8 * C;
  long total = (long)B * D2 * D2 * D2 * C8;
  if (idx >= total) return;
  int cc = (int)(idx % C8);
  long v = idx / C8;
  int xx = (int)(v % D2); v /= D2;
  int yy = (int)(v % D2); v /= D2;
  int zz = (int)(v % D2); v /= D2;
  int b = (int)v;
  int s8 = cc / C, c = cc % C;
  int Dd = 2 * D2;
  tmp[idx] = x[((((long)b * Dd + (2 * zz + od[s8])) * Dd + (2 * yy + oh[s8])) * Dd +
                (2 * xx + ow[s8])) * C + c];
}

__global__ void k_relidx(int* __restrict__ relidx) {
  int idx = blockIdx.x * blockDim.x + threadIdx.x;
  if (idx >= 343 * 343) return;
  int i = idx / 343, j = idx % 343;
  int d0 = i / 49, h0 = (i / 7) % 7, w0 = i % 7;
  int d1 = j / 49, h1 = (j / 7) % 7, w1 = j % 7;
  relidx[idx] = (d0 - d1 + 6) * 169 + (h0 - h1 + 6) * 13 + (w0 - w1 + 6);
}

// =============================== WMMA GEMM ===============================
// out[M,N] = A[M,K] * W[N,K]^T (+bias) (+gelu) (+resid). f16 operands, f32 accum.
// One wave per 16x16 tile, 8 waves / block. K % 8 == 0 guaranteed by caller
// (all K are multiples of 48), so 16-byte chunks are aligned and never straddle K.
__global__ void k_gemm_wmma(const _Float16* __restrict__ A, const _Float16* __restrict__ W,
                            const float* __restrict__ bias, const float* __restrict__ resid,
                            void* __restrict__ out, int M, int K, int N,
                            int act, int out_f16) {
  int lane = threadIdx.x & 31;
  int wave = threadIdx.x >> 5;
  int tiles_m = (M + 15) >> 4;
  int tiles_n = (N + 15) >> 4;
  long tile = (long)blockIdx.x * 8 + wave;
  if (tile >= (long)tiles_m * tiles_n) return;      // wave-uniform exit
  int tm = (int)(tile / tiles_n) * 16;
  int tn = (int)(tile % tiles_n) * 16;
  int am = tm + (lane & 15);
  int wn = tn + (lane & 15);
  int amc = (am < M) ? am : 0;                      // clamped rows; garbage rows never stored
  int wnc = (wn < N) ? wn : 0;
  const h8* Arow = (const h8*)(A + (long)amc * K);
  const h8* Wrow = (const h8*)(W + (long)wnc * K);
  int kch = K >> 3;                                 // # of 8-half chunks per row
  h8 z8 = {};
  v8f acc = {};
  for (int k0 = 0; k0 < K; k0 += 32) {
    int kc = k0 >> 3;
    // A chunks: lane<16 -> {kc, kc+2}; lane>=16 -> {kc+1, kc+3}
    int ca0 = kc + ((lane < 16) ? 0 : 1);
    int ca1 = ca0 + 2;
    h8 a0 = Arow[(ca0 < kch) ? ca0 : 0]; if (ca0 >= kch) a0 = z8;
    h8 a1 = Arow[(ca1 < kch) ? ca1 : 0]; if (ca1 >= kch) a1 = z8;
    // B chunks: lane<16 -> {kc, kc+2}; lane>=16 -> {kc+1, kc+3}
    int cb0 = kc + ((lane < 16) ? 0 : 1);
    int cb1 = cb0 + 2;
    h8 b0 = Wrow[(cb0 < kch) ? cb0 : 0]; if (cb0 >= kch) b0 = z8;
    h8 b1 = Wrow[(cb1 < kch) ? cb1 : 0]; if (cb1 >= kch) b1 = z8;
    v16h a = __builtin_shufflevector(a0, a1, 0, 1, 2, 3, 4, 5, 6, 7, 8, 9, 10, 11, 12, 13, 14, 15);
    v16h b = __builtin_shufflevector(b0, b1, 0, 1, 2, 3, 4, 5, 6, 7, 8, 9, 10, 11, 12, 13, 14, 15);
    acc = __builtin_amdgcn_wmma_f32_16x16x32_f16(false, a, false, b, (short)0, acc, false, false);
  }
  int col = tn + (lane & 15);
  int rbase = tm + ((lane < 16) ? 0 : 8);
#pragma unroll
  for (int r = 0; r < 8; ++r) {
    int row = rbase + r;
    if (row < M && col < N) {
      float v = acc[r];
      if (bias) v += bias[col];
      if (act == 1) v = 0.5f * v * (1.f + erff(v * 0.70710678118654752f));  // exact GELU
      if (resid) v += resid[(long)row * N + col];
      if (out_f16) ((_Float16*)out)[(long)row * N + col] = (_Float16)v;
      else         ((float*)out)[(long)row * N + col] = v;
    }
  }
}

// =============================== fused window attention ===============================
__device__ __forceinline__ int region_id(int c, int Sp, int wsz, int sh) {
  if (c < Sp - wsz) return 0;
  if (c < Sp - sh) return 1;
  return 2;
}

// one wave per (16-query tile, head, window). qkv f16: (bw*nseq, 3C); out f16: (bw*nseq, C). hd==16.
__global__ void k_attn(const _Float16* __restrict__ qkv, const float* __restrict__ rpb,
                       const int* __restrict__ relidx, _Float16* __restrict__ out,
                       int nseq, int npad, int C, int heads,
                       int wsz, int nwd, int Sp, int shift, int shifted) {
  __shared__ _Float16 Klds[ANMAX * 16];
  __shared__ _Float16 Vlds[ANMAX * 16];
  __shared__ float    Slds[16 * ANMAX];
  int lane = threadIdx.x;
  int tq = blockIdx.x;
  int h = blockIdx.y;
  int bw = blockIdx.z;
  int C3 = 3 * C;
  long tokbase = (long)bw * nseq;
  h8 z8 = {};

  // stage K,V into LDS as 16B chunks (2 chunks of 8 halves per token)
#ifdef USE_ASYNC_LDS
  for (int t = lane; t < nseq * 2; t += 32) {
    int n = t >> 1, cp = t & 1;
    const _Float16* gk = qkv + (tokbase + n) * C3 + C + h * 16 + cp * 8;
    __builtin_amdgcn_global_load_async_to_lds_b128(
        (__attribute__((address_space(1))) v4i*)gk,
        (__attribute__((address_space(3))) v4i*)&Klds[n * 16 + cp * 8], 0, 0);
    __builtin_amdgcn_global_load_async_to_lds_b128(
        (__attribute__((address_space(1))) v4i*)(gk + C),
        (__attribute__((address_space(3))) v4i*)&Vlds[n * 16 + cp * 8], 0, 0);
  }
#else
  for (int t = lane; t < nseq * 2; t += 32) {
    int n = t >> 1, cp = t & 1;
    const h8* gk = (const h8*)(qkv + (tokbase + n) * C3 + C + h * 16) + cp;
    const h8* gv = (const h8*)(qkv + (tokbase + n) * C3 + 2 * C + h * 16) + cp;
    *(h8*)&Klds[n * 16 + cp * 8] = *gk;
    *(h8*)&Vlds[n * 16 + cp * 8] = *gv;
  }
#endif
  for (int t = nseq * 2 + lane; t < npad * 2; t += 32) {  // zero pad tail tokens
    int n = t >> 1, cp = t & 1;
    *(h8*)&Klds[n * 16 + cp * 8] = z8;
    *(h8*)&Vlds[n * 16 + cp * 8] = z8;
  }

  // Q fragment: one 16B load, scaled by hd^-0.5 = 0.25 (exact in f16).
  // A-layout: lane<16 uses d=0..7 (chunk0), lane>=16 uses d=8..15 (chunk1); upper K half zero.
  int qm = tq * 16 + (lane & 15);
  int qmc = (qm < nseq) ? qm : 0;
  const h8* qrow = (const h8*)(qkv + (tokbase + qmc) * C3 + h * 16);
  h8 q0 = qrow[(lane < 16) ? 0 : 1];
  _Float16 qscale = (qm < nseq) ? (_Float16)0.25f : (_Float16)0.f;
  q0 = q0 * qscale;
  v16h aq = __builtin_shufflevector(q0, z8, 0, 1, 2, 3, 4, 5, 6, 7, 8, 9, 10, 11, 12, 13, 14, 15);

#ifdef USE_ASYNC_LDS
  __builtin_amdgcn_s_wait_asynccnt(0);
#endif
  __syncthreads();

  int nw = nwd * nwd * nwd;
  int wlin = bw % nw;
  int wz = wlin / (nwd * nwd), wy = (wlin / nwd) % nwd, wx = wlin % nwd;

  // S = Q K^T + bias (+shift mask), tile by tile
  for (int kt = 0; kt < npad / 16; ++kt) {
    // B-frag: key (kt*16+lane%16); lane<16 -> d 0..7 (chunk0), lane>=16 -> d 8..15 (chunk1); upper K half zero
    h8 b0 = z8, b1 = z8;
    {
      const h8* kr = (const h8*)&Klds[(kt * 16 + (lane & 15)) * 16];
      b0 = kr[(lane < 16) ? 0 : 1];
    }
    v16h bk = __builtin_shufflevector(b0, b1, 0, 1, 2, 3, 4, 5, 6, 7, 8, 9, 10, 11, 12, 13, 14, 15);
    v8f s = {};
    s = __builtin_amdgcn_wmma_f32_16x16x32_f16(false, aq, false, bk, (short)0, s, false, false);
    int col = kt * 16 + (lane & 15);
    int rbase = (lane < 16) ? 0 : 8;
#pragma unroll
    for (int r = 0; r < 8; ++r) {
      int row = rbase + r;
      int qrow = tq * 16 + row;
      float val = -1e30f;
      if (col < nseq) {
        val = s[r];
        if (qrow < nseq) {
          val += rpb[relidx[qrow * 343 + col] * heads + h];
          if (shifted) {
            int ezq = qrow / (wsz * wsz), eyq = (qrow / wsz) % wsz, exq = qrow % wsz;
            int ezk = col / (wsz * wsz), eyk = (col / wsz) % wsz, exk = col % wsz;
            int idq = region_id(wz * wsz + ezq, Sp, wsz, shift) * 9 +
                      region_id(wy * wsz + eyq, Sp, wsz, shift) * 3 +
                      region_id(wx * wsz + exq, Sp, wsz, shift);
            int idk = region_id(wz * wsz + ezk, Sp, wsz, shift) * 9 +
                      region_id(wy * wsz + eyk, Sp, wsz, shift) * 3 +
                      region_id(wx * wsz + exk, Sp, wsz, shift);
            if (idq != idk) val -= 100.f;
          }
        }
      }
      Slds[row * ANMAX + col] = val;
    }
  }
  __syncthreads();

  // softmax rows (lanes 0..15, one row each)
  if (lane < 16) {
    int row = lane;
    float mx = -1e30f;
    for (int c = 0; c < nseq; ++c) mx = fmaxf(mx, Slds[row * ANMAX + c]);
    float sum = 0.f;
    for (int c = 0; c < nseq; ++c) {
      float e = expf(Slds[row * ANMAX + c] - mx);
      Slds[row * ANMAX + c] = e;
      sum += e;
    }
    float inv = 1.f / sum;
    for (int c = 0; c < npad; ++c)
      Slds[row * ANMAX + c] = (c < nseq) ? Slds[row * ANMAX + c] * inv : 0.f;
  }
  __syncthreads();

  // O = P * V
  v8f o = {};
  for (int k0 = 0; k0 < npad; k0 += 32) {
    v16h ap, bv;
#pragma unroll
    for (int j = 0; j < 16; ++j) {
      int kk = a_kidx(lane, j);
      ap[j] = (_Float16)Slds[(lane & 15) * ANMAX + k0 + kk];
      int kb = kk;
      bv[j] = Vlds[(k0 + kb) * 16 + (lane & 15)];
    }
    o = __builtin_amdgcn_wmma_f32_16x16x32_f16(false, ap, false, bv, (short)0, o, false, false);
  }
  int dcol = lane & 15;
  int rbase = (lane < 16) ? 0 : 8;
#pragma unroll
  for (int r = 0; r < 8; ++r) {
    int qrow = tq * 16 + rbase + r;
    if (qrow < nseq)
      out[(tokbase + qrow) * C + h * 16 + dcol] = (_Float16)o[r];
  }
}

// =============================== host orchestration ===============================

static void launch_gemm(const _Float16* A, const _Float16* W, const float* bias,
                        const float* resid, void* out, int M, int K, int N,
                        int act, int out_f16, hipStream_t stream) {
  long tiles = (long)((M + 15) / 16) * ((N + 15) / 16);
  unsigned blocks = (unsigned)((tiles + 7) / 8);
  k_gemm_wmma<<<blocks, 256, 0, stream>>>(A, W, bias, resid, out, M, K, N, act, out_f16);
}

extern "C" void kernel_launch(void* const* d_in, const int* in_sizes, int n_in,
                              void* d_out, int out_size, void* d_ws, size_t ws_size,
                              hipStream_t stream) {
  (void)in_sizes; (void)n_in; (void)out_size; (void)ws_size;
  const float* x_in    = (const float*)d_in[0];
  const float* patch_w = (const float*)d_in[1];
  const float* patch_b = (const float*)d_in[2];
  float* out = (float*)d_out;

  char* wsb = (char*)d_ws;
  size_t off = 0;
  auto alloc = [&](size_t bytes) -> void* {
    void* p = wsb + off;
    off += (bytes + 255) & ~(size_t)255;
    return p;
  };
  int*       relidx  = (int*)alloc(343 * 343 * sizeof(int));
  float*     bufX    = (float*)alloc((size_t)3145728 * 4);   // activation (f32 master)
  float*     bufPROJ = (float*)alloc((size_t)4116000 * 4);   // proj out (f32, pre-residual)
  float*     bufTMP  = (float*)alloc((size_t)3145728 * 4);   // merge concat (f32)
  _Float16*  bufXNh  = (_Float16*)alloc((size_t)3145728 * 2);  // LN out (f16)
  _Float16*  bufWINh = (_Float16*)alloc((size_t)4116000 * 2);  // windowed tokens (f16)
  _Float16*  bufQKVh = (_Float16*)alloc((size_t)12582912 * 2); // qkv / aliased MLP hidden (f16)
  _Float16*  bufATTh = (_Float16*)alloc((size_t)4116000 * 2);  // attention out (f16)
  _Float16*  bufWh   = (_Float16*)alloc((size_t)2359296 * 2);  // converted weight (f16)
  _Float16*  bufMLPh = bufQKVh;

  auto cvtw = [&](const float* w, long n) -> const _Float16* {
    k_cvt_f16<<<(unsigned)((n + 255) / 256), 256, 0, stream>>>(w, bufWh, n);
    return (const _Float16*)bufWh;
  };

  k_relidx<<<(343 * 343 + 255) / 256, 256, 0, stream>>>(relidx);

  // patch embed -> channels-last (2,32,32,32,48) f32
  {
    long total = 2L * 32 * 32 * 32 * 48;
    k_patch_embed<<<(unsigned)((total + 255) / 256), 256, 0, stream>>>(
        x_in, patch_w, patch_b, bufX, 2, 32, 32, 32, 48);
  }
  long out_off = 0;
  {
    long nvox = 2L * 32 * 32 * 32;
    k_proj_norm_out<<<(unsigned)((nvox + 255) / 256), 256, 0, stream>>>(
        bufX, out + out_off, 2, 48, 32, 32, 32);
    out_off += nvox * 48;
  }

  const int HEADS_[4] = {3, 6, 12, 24};
  for (int s = 0; s < 4; ++s) {
    int D = 32 >> s;
    int C = 48 << s;
    int heads = HEADS_[s];
    long tokens = 2L * D * D * D;
    int base = 3 + s * 16;
    const float* n1g = (const float*)d_in[base + 0];
    const float* n1b = (const float*)d_in[base + 1];
    const float* qkvw = (const float*)d_in[base + 2];
    const float* qkvb = (const float*)d_in[base + 3];
    const float* rpb  = (const float*)d_in[base + 4];
    const float* pw   = (const float*)d_in[base + 5];
    const float* pb   = (const float*)d_in[base + 6];
    const float* n2g  = (const float*)d_in[base + 7];
    const float* n2b  = (const float*)d_in[base + 8];
    const float* f1w  = (const float*)d_in[base + 9];
    const float* f1b  = (const float*)d_in[base + 10];
    const float* f2w  = (const float*)d_in[base + 11];
    const float* f2b  = (const float*)d_in[base + 12];
    const float* mng  = (const float*)d_in[base + 13];
    const float* mnb  = (const float*)d_in[base + 14];
    const float* mw   = (const float*)d_in[base + 15];

    int wsz = (D <= 7) ? D : 7;                 // get_window_size clamp
    int Sp = ((D + wsz - 1) / wsz) * wsz;       // padded extent
    int nwd = Sp / wsz;
    int nseq = wsz * wsz * wsz;
    int npad = (nseq + 31) & ~31;
    int nw = nwd * nwd * nwd;
    long tokw = 2L * nw * nseq;

    for (int blk = 0; blk < 2; ++blk) {
      int shift = (blk == 1 && D > 7) ? 3 : 0;  // clamp also kills the shift
      int shifted = (shift != 0) ? 1 : 0;

      k_layernorm<<<(unsigned)((tokens + 255) / 256), 256, 0, stream>>>(
          bufX, n1g + blk * C, n1b + blk * C, bufXNh, tokens, C);

      { long tot = tokw * C;
        k_win_gather<<<(unsigned)((tot + 255) / 256), 256, 0, stream>>>(
            bufXNh, bufWINh, 2, D, C, wsz, nwd, Sp, shift); }

      launch_gemm(bufWINh, cvtw(qkvw + (long)blk * 3 * C * C, 3L * C * C),
                  qkvb + blk * 3 * C, nullptr, bufQKVh, (int)tokw, C, 3 * C,
                  0, 1, stream);

      { dim3 g((nseq + 15) / 16, heads, 2 * nw);
        k_attn<<<g, 32, 0, stream>>>(bufQKVh, rpb + (long)blk * 2197 * heads, relidx, bufATTh,
                                     nseq, npad, C, heads, wsz, nwd, Sp, shift, shifted); }

      launch_gemm(bufATTh, cvtw(pw + (long)blk * C * C, (long)C * C),
                  pb + blk * C, nullptr, bufPROJ, (int)tokw, C, C, 0, 0, stream);

      { long tot = tokens * C;
        k_win_scatter_add<<<(unsigned)((tot + 255) / 256), 256, 0, stream>>>(
            bufPROJ, bufX, 2, D, C, wsz, nwd, Sp, shift); }

      k_layernorm<<<(unsigned)((tokens + 255) / 256), 256, 0, stream>>>(
          bufX, n2g + blk * C, n2b + blk * C, bufXNh, tokens, C);

      launch_gemm(bufXNh, cvtw(f1w + (long)blk * 4 * C * C, 4L * C * C),
                  f1b + blk * 4 * C, nullptr, bufMLPh, (int)tokens, C, 4 * C,
                  1, 1, stream);                                   // + exact GELU, f16 out

      launch_gemm(bufMLPh, cvtw(f2w + (long)blk * 4 * C * C, 4L * C * C),
                  f2b + blk * C, bufX, bufX, (int)tokens, 4 * C, C,
                  0, 0, stream);                                   // + residual, f32 out
    }

    // patch merging
    int D2 = D / 2;
    long tok2 = 2L * D2 * D2 * D2;
    { long tot = tok2 * 8 * C;
      k_merge_gather<<<(unsigned)((tot + 255) / 256), 256, 0, stream>>>(bufX, bufTMP, 2, D2, C); }
    k_layernorm<<<(unsigned)((tok2 + 255) / 256), 256, 0, stream>>>(
        bufTMP, mng, mnb, bufXNh, tok2, 8 * C);
    launch_gemm(bufXNh, cvtw(mw, 16L * C * C), nullptr, nullptr, bufX,
                (int)tok2, 8 * C, 2 * C, 0, 0, stream);

    k_proj_norm_out<<<(unsigned)((tok2 + 255) / 256), 256, 0, stream>>>(
        bufX, out + out_off, 2, 2 * C, D2, D2, D2);
    out_off += tok2 * 2 * C;
  }
}
